// Attention_56307021251085
// MI455X (gfx1250) — matrix operand order repaired and checked by reading, not compiled, not run on hardware
//
#include <hip/hip_runtime.h>

// ---------------------------------------------------------------------------
// MI455X (gfx1250) fused multi-head causal attention, bf16 WMMA pipeline.
//   1) cvt_x_k       : x fp32 -> bf16
//   2) build_wproj_k : pack Q/K/V weights -> Wt[N=3072][K=1024] bf16
//   3) build_wcat_k  : pack W            -> Wt[N=1024][K=1024] bf16
//   4) gemm_bf16_k   : qkv = xb @ Wproj   (bf16 out)           [v_wmma]
//   5) flash_attn_k  : online-softmax causal attention         [v_wmma]
//   6) gemm_bf16_k   : z = (Y @ Wcat) * 0.0625 (fp32 out)      [v_wmma]
// Tile staging uses GLOBAL_LOAD_ASYNC_TO_LDS_B128 (ASYNCcnt DMA path).
// Workspace use: ~88 MB (xb 16M, wproj 6M, qkv 48M, wcat 2M, y 16M).
// ---------------------------------------------------------------------------

typedef __bf16 bf16_t;
typedef __attribute__((ext_vector_type(16))) __bf16 bfx16;
typedef __attribute__((ext_vector_type(8)))  __bf16 bfx8;
typedef __attribute__((ext_vector_type(2)))  __bf16 bfx2;
typedef __attribute__((ext_vector_type(8)))  float  fx8;

#define B_   4
#define C_   2048
#define D_   1024
#define H_   16
#define DQ_  64
#define ROWS_ (B_ * C_)      // 8192
#define QKV_N (3 * H_ * DQ_) // 3072

#define WMMA_BF16(a, b, c) \
  __builtin_amdgcn_wmma_f32_16x16x32_bf16(false, (a), false, (b), (short)0, (c), false, false)

// ---- async global->LDS copy (CDNA5 DMA path), guarded so compile never breaks
#if defined(__has_builtin)
#if __has_builtin(__builtin_amdgcn_global_load_async_to_lds_b128)
#define USE_ASYNC_LDS 1
#endif
#if __has_builtin(__builtin_amdgcn_s_wait_asynccnt)
#define HAVE_WAIT_ASYNC 1
#endif
#endif
#ifndef USE_ASYNC_LDS
#define USE_ASYNC_LDS 0
#endif
#ifndef HAVE_WAIT_ASYNC
#define HAVE_WAIT_ASYNC 0
#endif

// Builtin signature (from the round-2 diagnostic): params are pointers to a
// GCC-style int4 vector in AS(1) / AS(3), plus two imm ints (offset, cpol).
typedef int v4i_t __attribute__((vector_size(4 * sizeof(int))));
#if USE_ASYNC_LDS
typedef __attribute__((address_space(1))) v4i_t* g_v4i_p;
typedef __attribute__((address_space(3))) v4i_t* l_v4i_p;
#endif

static __device__ __forceinline__ void cp16_g2s(bf16_t* dst_lds, const bf16_t* src_g) {
#if USE_ASYNC_LDS
  // AS1 from flat address via inttoptr (global flat addr == AS1 addr);
  // AS3 via C-style addrspace cast of the __shared__-derived pointer.
  __builtin_amdgcn_global_load_async_to_lds_b128(
      (g_v4i_p)(unsigned long long)(const void*)src_g, (l_v4i_p)dst_lds, 0, 0);
#else
  *(bfx8*)dst_lds = *(const bfx8*)src_g;
#endif
}
static __device__ __forceinline__ void cp_wait() {
#if USE_ASYNC_LDS
#if HAVE_WAIT_ASYNC
  __builtin_amdgcn_s_wait_asynccnt(0);
#else
  asm volatile("s_wait_asynccnt 0x0" ::: "memory");
#endif
#endif
}

static __device__ __forceinline__ bfx16 join16(bfx8 lo, bfx8 hi) {
  return __builtin_shufflevector(lo, hi, 0,1,2,3,4,5,6,7,8,9,10,11,12,13,14,15);
}
// A-matrix 16x32 bf16 fragment: lane m=lane&15, elem j -> K = j + 8*half (+8 for j>=8)
static __device__ __forceinline__ bfx16 frag_A(const bf16_t* rowp, int lh) {
  bfx8 lo = *(const bfx8*)(rowp + 8 * lh);
  bfx8 hi = *(const bfx8*)(rowp + 16 + 8 * lh);
  return join16(lo, hi);
}
// B-matrix 32x16 bf16 fragment: lane n=lane&15, elem j -> K = j + 8*half (+8 for j>=8)
static __device__ __forceinline__ bfx16 frag_B(const bf16_t* rowp, int lh) {
  bfx8 lo = *(const bfx8*)(rowp + 8 * lh);
  bfx8 hi = *(const bfx8*)(rowp + 16 + 8 * lh);
  return join16(lo, hi);
}

// ---------------------------------------------------------------------------
// Tiled bf16 GEMM: out[M][N] = A[M][K=1024] * Bt[N][K=1024]^T, K hardcoded.
// Block 256 thr = 8 waves (4x2), block tile 128x128, K-step 32, double buffer.
// ---------------------------------------------------------------------------
template <bool F32OUT>
__global__ __launch_bounds__(256) void gemm_bf16_k(
    const bf16_t* __restrict__ A, const bf16_t* __restrict__ Bt,
    void* __restrict__ Cout, int N, float scale) {
  constexpr int K  = 1024;
  constexpr int LR = 40;  // 32 + 8 pad elements (80 B: 16B-aligned, bank-spread)
  __shared__ __attribute__((aligned(16))) bf16_t As[2][128 * LR];
  __shared__ __attribute__((aligned(16))) bf16_t Bs[2][128 * LR];
  const int tid = threadIdx.x;
  const int wid = tid >> 5, lane = tid & 31, lh = lane >> 4, l16 = lane & 15;
  const int wM = wid >> 1, wN = wid & 1;
  const size_t m0 = (size_t)blockIdx.x * 128;
  const size_t n0 = (size_t)blockIdx.y * 128;

  fx8 acc[2][4] = {};

  auto load_tiles = [&](int k0, int buf) {
#pragma unroll
    for (int i = 0; i < 2; ++i) {
      int c = tid * 2 + i;           // 512 chunks of 16B per 128x32 tile
      int row = c >> 2, kc = (c & 3) * 8;
      cp16_g2s(&As[buf][row * LR + kc], A + (m0 + row) * K + k0 + kc);
      cp16_g2s(&Bs[buf][row * LR + kc], Bt + (n0 + row) * K + k0 + kc);
    }
  };
  load_tiles(0, 0);
  cp_wait();
  __syncthreads();
#pragma unroll 1
  for (int step = 0; step < K / 32; ++step) {
    const int buf = step & 1;
    if (step + 1 < K / 32) load_tiles((step + 1) * 32, buf ^ 1);  // DMA next tile
    bfx16 af[2], bfr[4];
#pragma unroll
    for (int ms = 0; ms < 2; ++ms)
      af[ms] = frag_A(&As[buf][(wM * 32 + ms * 16 + l16) * LR], lh);
#pragma unroll
    for (int ns = 0; ns < 4; ++ns)
      bfr[ns] = frag_B(&Bs[buf][(wN * 64 + ns * 16 + l16) * LR], lh);
#pragma unroll
    for (int ms = 0; ms < 2; ++ms)
#pragma unroll
      for (int ns = 0; ns < 4; ++ns)
        acc[ms][ns] = WMMA_BF16(af[ms], bfr[ns], acc[ms][ns]);
    cp_wait();        // DMA for next buffer complete before anyone reads it
    __syncthreads();
  }
  // C/D layout: lane n = lane&15, VGPR r -> row r + 8*half
#pragma unroll
  for (int ms = 0; ms < 2; ++ms)
#pragma unroll
    for (int ns = 0; ns < 4; ++ns) {
      const size_t col = n0 + wN * 64 + ns * 16 + l16;
#pragma unroll
      for (int r = 0; r < 8; ++r) {
        const size_t row = m0 + wM * 32 + ms * 16 + r + 8 * lh;
        const float v = acc[ms][ns][r] * scale;
        if (F32OUT) ((float*)Cout)[row * (size_t)N + col] = v;
        else        ((bf16_t*)Cout)[row * (size_t)N + col] = (bf16_t)v;
      }
    }
}

// ---------------------------------------------------------------------------
// Flash attention: block = (qtile 64 rows, head, batch); 128 thr = 4 waves,
// each wave owns 16 query rows. 64-key tiles staged in LDS.
// qkv layout: [B][C][3][H][64] bf16 (row stride 3072).
// ---------------------------------------------------------------------------
__global__ __launch_bounds__(128) void flash_attn_k(
    const bf16_t* __restrict__ qkv, bf16_t* __restrict__ Y) {
  constexpr int RS = QKV_N;  // 3072
  constexpr int LR = 72;     // 64 + 8 pad (144 B rows)
  __shared__ __attribute__((aligned(16))) bf16_t Kt[64 * LR];      // [key][dq]
  __shared__ __attribute__((aligned(16))) bf16_t Vt[64 * LR];      // [dv][key]
  __shared__ __attribute__((aligned(16))) bf16_t Pl[4][16 * LR];   // per-wave P
  const int tid = threadIdx.x;
  const int wid = tid >> 5, lane = tid & 31, lh = lane >> 4, l16 = lane & 15;
  const int qt = blockIdx.x, h = blockIdx.y, b = blockIdx.z;
  const int qb = qt * 64;
  const size_t base = (size_t)b * C_ * RS;
  const float CEXP = 1.4426950408889634f / 1024.0f;  // (sq^2/dq)*log2(e), sq folded

  // Q fragments (A-matrix) straight from global: dq contiguous per row.
  const bf16_t* qrowp = qkv + base + (size_t)(qb + wid * 16 + l16) * RS + h * 64;
  bfx16 qf[2];
#pragma unroll
  for (int ks = 0; ks < 2; ++ks) qf[ks] = frag_A(qrowp + ks * 32, lh);

  fx8 o[4] = {};
  float mrow[8], lrow[8];
#pragma unroll
  for (int r = 0; r < 8; ++r) { mrow[r] = -3.0e38f; lrow[r] = 0.f; }

  for (int kt = 0; kt <= qt; ++kt) {
    const int k0 = kt * 64;
    const bf16_t* kp = qkv + base + (size_t)k0 * RS + H_ * 64 + h * 64;
    const bf16_t* vp = qkv + base + (size_t)k0 * RS + 2 * H_ * 64 + h * 64;
    // K tile: 64x64, row-major into LDS via async DMA (coalesced 16B chunks).
#pragma unroll
    for (int i = 0; i < 4; ++i) {
      int c = tid + i * 128;
      int row = c >> 3, kc = (c & 7) * 8;
      cp16_g2s(&Kt[row * LR + kc], kp + (size_t)row * RS + kc);
    }
    // V tile transposed into LDS: read 2 key-rows x 8 dv, write bfx2 pairs.
#pragma unroll
    for (int i = 0; i < 2; ++i) {
      int s = tid + i * 128;
      int k2 = (s >> 3) * 2, dvc = (s & 7) * 8;
      bfx8 a0 = *(const bfx8*)(vp + (size_t)k2 * RS + dvc);
      bfx8 a1 = *(const bfx8*)(vp + (size_t)(k2 + 1) * RS + dvc);
#pragma unroll
      for (int e = 0; e < 8; ++e) {
        bfx2 pair = {a0[e], a1[e]};
        *(bfx2*)&Vt[(dvc + e) * LR + k2] = pair;
      }
    }
    cp_wait();
    __syncthreads();

    // S = Q @ K^T : 16 rows x 64 keys per wave (4 N-tiles x 2 K-steps).
    fx8 s4[4] = {};
#pragma unroll
    for (int ns = 0; ns < 4; ++ns)
#pragma unroll
      for (int ks = 0; ks < 2; ++ks)
        s4[ns] = WMMA_BF16(qf[ks], frag_B(&Kt[(ns * 16 + l16) * LR + ks * 32], lh), s4[ns]);

    if (kt == qt) {  // diagonal tile: mask key > row
#pragma unroll
      for (int ns = 0; ns < 4; ++ns) {
        const int key = ns * 16 + l16;
#pragma unroll
        for (int r = 0; r < 8; ++r)
          if (key > wid * 16 + r + 8 * lh) s4[ns][r] = -3.0e38f;
      }
    }

    // Online softmax: each row lives in one 16-lane half at a fixed VGPR idx.
    float mnew[8], alpha[8], lacc[8];
#pragma unroll
    for (int r = 0; r < 8; ++r) {
      float mx = fmaxf(fmaxf(s4[0][r], s4[1][r]), fmaxf(s4[2][r], s4[3][r]));
#pragma unroll
      for (int off = 8; off >= 1; off >>= 1) mx = fmaxf(mx, __shfl_xor(mx, off, 32));
      mnew[r]  = fmaxf(mrow[r], mx);
      alpha[r] = exp2f((mrow[r] - mnew[r]) * CEXP);
      lacc[r]  = 0.f;
    }
    bf16_t* pw = &Pl[wid][0];
#pragma unroll
    for (int ns = 0; ns < 4; ++ns)
#pragma unroll
      for (int r = 0; r < 8; ++r) {
        const float pv = exp2f((s4[ns][r] - mnew[r]) * CEXP);
        lacc[r] += pv;
        pw[(r + 8 * lh) * LR + ns * 16 + l16] = (bf16_t)pv;  // C-layout -> [row][key]
      }
#pragma unroll
    for (int r = 0; r < 8; ++r) {
      float t = lacc[r];
#pragma unroll
      for (int off = 8; off >= 1; off >>= 1) t += __shfl_xor(t, off, 32);
      lrow[r] = lrow[r] * alpha[r] + t;
      mrow[r] = mnew[r];
    }
#pragma unroll
    for (int ns = 0; ns < 4; ++ns)
#pragma unroll
      for (int r = 0; r < 8; ++r) o[ns][r] *= alpha[r];

    // O += P @ V : reload P as A-fragments, Vt rows give B-fragments.
    bfx16 pf[2];
#pragma unroll
    for (int ks = 0; ks < 2; ++ks) pf[ks] = frag_A(&Pl[wid][l16 * LR + ks * 32], lh);
#pragma unroll
    for (int ns = 0; ns < 4; ++ns)
#pragma unroll
      for (int ks = 0; ks < 2; ++ks)
        o[ns] = WMMA_BF16(pf[ks], frag_B(&Vt[(ns * 16 + l16) * LR + ks * 32], lh), o[ns]);
    __syncthreads();
  }
  // y[b,c,h*64+dv] = O / l  (sv and final scale folded into last GEMM)
#pragma unroll
  for (int r = 0; r < 8; ++r) {
    const float inv = 1.0f / lrow[r];
    const size_t row = (size_t)b * C_ + qb + wid * 16 + r + 8 * lh;
#pragma unroll
    for (int ns = 0; ns < 4; ++ns)
      Y[row * 1024 + h * 64 + ns * 16 + l16] = (bf16_t)(o[ns][r] * inv);
  }
}

// ------------------------------- prep kernels -------------------------------
__global__ __launch_bounds__(256) void cvt_x_k(const float* __restrict__ x,
                                               bf16_t* __restrict__ xb) {
  const size_t i = ((size_t)blockIdx.x * 256 + threadIdx.x) * 4;
#pragma unroll
  for (int j = 0; j < 4; ++j) xb[i + j] = (bf16_t)x[i + j];
}

// Wt[n][k], n = m3*1024 + h*64 + e, value = {Q,K,V}[e, k, h]; src shape (64,1024,16)
__global__ __launch_bounds__(256) void build_wproj_k(
    const float* __restrict__ Q, const float* __restrict__ Km,
    const float* __restrict__ V, bf16_t* __restrict__ wt) {
  const size_t idx = (size_t)blockIdx.x * 256 + threadIdx.x;  // < 3072*1024
  const int n = (int)(idx >> 10), k = (int)(idx & 1023);
  const int m3 = n >> 10, rem = n & 1023, h = rem >> 6, e = rem & 63;
  const float* src = (m3 == 0) ? Q : ((m3 == 1) ? Km : V);
  wt[idx] = (bf16_t)src[(size_t)e * (1024 * 16) + (size_t)k * 16 + h];
}

// Wt[n=d][k=h*64+v] = W[d, v, h]; W shape (1024, 64, 16)
__global__ __launch_bounds__(256) void build_wcat_k(const float* __restrict__ W,
                                                    bf16_t* __restrict__ wt) {
  const size_t idx = (size_t)blockIdx.x * 256 + threadIdx.x;  // < 1024*1024
  const int d = (int)(idx >> 10), k = (int)(idx & 1023);
  const int h = k >> 6, v = k & 63;
  wt[idx] = (bf16_t)W[(size_t)d * 1024 + v * 16 + h];
}

// ------------------------------- launcher -----------------------------------
extern "C" void kernel_launch(void* const* d_in, const int* in_sizes, int n_in,
                              void* d_out, int out_size, void* d_ws, size_t ws_size,
                              hipStream_t stream) {
  (void)in_sizes; (void)n_in; (void)out_size; (void)ws_size;
  const float* x = (const float*)d_in[0];
  const float* Q = (const float*)d_in[1];
  const float* K = (const float*)d_in[2];
  const float* V = (const float*)d_in[3];
  const float* W = (const float*)d_in[4];

  char* ws = (char*)d_ws;
  bf16_t* xb    = (bf16_t*)(ws);                 // 8192*1024      = 16 MB
  bf16_t* wproj = (bf16_t*)(ws + (16u << 20));   // 3072*1024      =  6 MB
  bf16_t* qkv   = (bf16_t*)(ws + (22u << 20));   // 8192*3072      = 48 MB
  bf16_t* wcat  = (bf16_t*)(ws + (70u << 20));   // 1024*1024      =  2 MB
  bf16_t* yb    = (bf16_t*)(ws + (72u << 20));   // 8192*1024      = 16 MB

  cvt_x_k<<<(ROWS_ * D_) / 1024, 256, 0, stream>>>(x, xb);
  build_wproj_k<<<(QKV_N * 1024) / 256, 256, 0, stream>>>(Q, K, V, wproj);
  build_wcat_k<<<(1024 * 1024) / 256, 256, 0, stream>>>(W, wcat);

  // qkv = xb @ Wproj (unscaled; sq folded into softmax scale, sv into final)
  gemm_bf16_k<false><<<dim3(ROWS_ / 128, QKV_N / 128), 256, 0, stream>>>(
      xb, wproj, (void*)qkv, QKV_N, 1.0f);

  flash_attn_k<<<dim3(C_ / 64, H_, B_), 128, 0, stream>>>(qkv, yb);

  // z = (Y @ Wcat) * (sv * sqrt(D/DV)/H) = * 0.0625, fp32 out
  gemm_bf16_k<true><<<dim3(ROWS_ / 128, D_ / 128), 256, 0, stream>>>(
      yb, wcat, d_out, D_, 0.0625f);
}
